// CorticalRegion_30794915512831
// MI455X (gfx1250) — matrix operand order repaired and checked by reading, not compiled, not run on hardware
//
#include <hip/hip_runtime.h>
#include <hip/hip_bf16.h>

// ---- problem constants (match reference) ----
#define Bc 64
#define Cc 256
#define Dc 256
#define Nn 512
#define KW 24
#define FEEDBACK 0.3f

// ---- LDS row strides (padded to avoid 64-bank conflicts) ----
#define SXS 264   // __bf16 stride for staged inputs   (132 dwords, 132%64=4)
#define DRS 520   // float  stride for drive buffer    (520%64=8)
#define SAS 520   // __bf16 stride for activation buf  (260 dwords, 260%64=4)

typedef __attribute__((ext_vector_type(16))) __bf16 v16bf;
typedef __attribute__((ext_vector_type(8)))  float  v8f;

__global__ __launch_bounds__(256)
void cortical_region_kernel(const float* __restrict__ x_input,
                            const float* __restrict__ x_context,
                            const float* __restrict__ W_ff,
                            const float* __restrict__ W_ctx,
                            const float* __restrict__ W_pred,
                            const float* __restrict__ bias,
                            const float* __restrict__ avg_activity,
                            float* __restrict__ out_act,
                            float* __restrict__ out_pred,
                            float* __restrict__ out_err) {
  extern __shared__ char smem[];
  // sAct aliases the staged-input region: sX/sXc are dead after phase 1,
  // sAct is only written after the phase-1 barrier.
  __bf16* sX   = (__bf16*)smem;                 // [64][SXS] bf16 x_input
  __bf16* sXc  = sX + Bc * SXS;                 // [64][SXS] bf16 0.3*x_context
  __bf16* sAct = (__bf16*)smem;                 // [64][SAS] activations (bf16)
  float*  sDr  = (float*)(sXc + Bc * SXS);      // [64][DRS] drive (f32)
  float*  sBo  = sDr + Bc * DRS;                // [512]     homeostatic boost

  const int c    = blockIdx.x;
  const int tid  = threadIdx.x;
  const int wave = tid >> 5;
  const int lane = tid & 31;
  const int half = lane >> 4;
  const int l16  = lane & 15;

  // ---------- stage inputs (f32 -> bf16; fold FEEDBACK into Xc) ----------
  for (int idx = tid; idx < Bc * Dc; idx += 256) {
    const int r = idx >> 8, col = idx & (Dc - 1);
    sX [r * SXS + col] = (__bf16)x_input[idx];
    sXc[r * SXS + col] = (__bf16)(FEEDBACK * x_context[idx]);
  }
  for (int n = tid; n < Nn; n += 256)  // HOMEOSTASIS*100 == 1.0 folded in
    sBo[n] = log1pf(0.05f / (avg_activity[(size_t)c * Nn + n] + 1e-6f));
  __syncthreads();

  const float* Wf = W_ff  + (size_t)c * Dc * Nn;
  const float* Wx = W_ctx + (size_t)c * Dc * Nn;
  const v8f vzero = {0.f, 0.f, 0.f, 0.f, 0.f, 0.f, 0.f, 0.f};

  // ---------- phase 1: drive = X*Wff + (0.3*Xc)*Wctx + bias ----------
  // wave owns N strip [wave*64, wave*64+64): acc[nt][r] = 16 16x16 tiles.
  // A fragments depend only on (ks, r): build once per K-step, reuse for 4 nt.
  {
    v8f acc[4][4];
    #pragma unroll
    for (int nt = 0; nt < 4; ++nt)
      #pragma unroll
      for (int r = 0; r < 4; ++r) acc[nt][r] = vzero;

    for (int ks = 0; ks < Dc / 32; ++ks) {
      const int kb = ks * 32;
      if (kb + 32 < Dc) {  // prefetch next K-step of streamed weights
        __builtin_prefetch(&Wf[(size_t)(kb + 32 + half * 16) * Nn + wave * 64], 0, 1);
        __builtin_prefetch(&Wx[(size_t)(kb + 32 + half * 16) * Nn + wave * 64], 0, 1);
      }
      v16bf aX[4], aC[4];  // A fragment: lane holds row M=l16, K per ISA layout
      #pragma unroll
      for (int r = 0; r < 4; ++r) {
        const int row = r * 16 + l16;
        #pragma unroll
        for (int j = 0; j < 16; ++j) {
          const int kk = kb + ((j < 8) ? j : j + 8) + half * 8;
          aX[r][j] = sX [row * SXS + kk];
          aC[r][j] = sXc[row * SXS + kk];
        }
      }
      #pragma unroll
      for (int nt = 0; nt < 4; ++nt) {
        const int nb = (wave * 4 + nt) * 16;
        v16bf bff, bcx;  // B fragment: lane holds col n=l16, K per ISA layout
        #pragma unroll
        for (int j = 0; j < 16; ++j) {
          const size_t kk = (size_t)(kb + ((j < 8) ? j : j + 8) + half * 8);
          bff[j] = (__bf16)Wf[kk * Nn + nb + l16];
          bcx[j] = (__bf16)Wx[kk * Nn + nb + l16];
        }
        #pragma unroll
        for (int r = 0; r < 4; ++r) {
          acc[nt][r] = __builtin_amdgcn_wmma_f32_16x16x32_bf16(
              false, aX[r], false, bff, (short)0, acc[nt][r], false, false);
          acc[nt][r] = __builtin_amdgcn_wmma_f32_16x16x32_bf16(
              false, aC[r], false, bcx, (short)0, acc[nt][r], false, false);
        }
      }
    }
    #pragma unroll
    for (int nt = 0; nt < 4; ++nt) {
      const int nb = (wave * 4 + nt) * 16;
      const float bv = bias[(size_t)c * Nn + nb + l16];
      #pragma unroll
      for (int r = 0; r < 4; ++r) {
        #pragma unroll
        for (int v = 0; v < 8; ++v)  // C/D layout: VGPR v -> M = v + 8*half, N = l16
          sDr[(r * 16 + v + 8 * half) * DRS + nb + l16] = acc[nt][r][v] + bv;
      }
    }
  }
  __syncthreads();

  // ---------- phase 2: boosted top-K (K=24) per (b,c) row of N=512 ----------
  for (int row = wave; row < Bc; row += 8) {
    float dv[16], bvv[16];
    #pragma unroll
    for (int i = 0; i < 16; ++i) {
      const int n = lane + 32 * i;
      const float d = sDr[row * DRS + n];
      dv[i]  = d;
      bvv[i] = d + sBo[n];
    }
    unsigned dead = 0u;    // bitmask of already-extracted local slots
    float thr = 0.f;
    for (int t = 0; t < KW; ++t) {
      float lm = -3.4e38f; int li = 0;
      #pragma unroll
      for (int i = 0; i < 16; ++i) {
        const float x = ((dead >> i) & 1u) ? -3.4e38f : bvv[i];
        if (x > lm) { lm = x; li = i; }
      }
      float m = lm;
      #pragma unroll
      for (int off = 16; off > 0; off >>= 1)
        m = fmaxf(m, __shfl_xor(m, off, 32));
      thr = m;  // after 24 extractions this is the 24th-largest
      const unsigned long long bal = __ballot(lm == m);
      const int src = (int)__ffsll((unsigned long long)bal) - 1;
      if (lane == src) dead |= (1u << li);
    }
    float s = 0.f, av[16];
    #pragma unroll
    for (int i = 0; i < 16; ++i) {
      const float a = (bvv[i] >= thr) ? fmaxf(dv[i], 0.f) : 0.f;
      av[i] = a; s += a;
    }
    #pragma unroll
    for (int off = 16; off > 0; off >>= 1)
      s += __shfl_xor(s, off, 32);
    const float sc = (float)KW / (s + 1e-8f);
    #pragma unroll
    for (int i = 0; i < 16; ++i) {
      const int n = lane + 32 * i;
      const float a = av[i] * sc;
      sAct[row * SAS + n] = (__bf16)a;   // bf16 copy for phase-3 A fragments
      out_act[(size_t)row * Cc * Nn + (size_t)c * Nn + n] = a;
    }
  }
  __syncthreads();

  // ---------- phase 3: pred = act * W_pred ; err = x - pred ----------
  const float* Wp = W_pred + (size_t)c * Nn * Dc;
  {
    v8f acc[2][4];
    #pragma unroll
    for (int nt = 0; nt < 2; ++nt)
      #pragma unroll
      for (int r = 0; r < 4; ++r) acc[nt][r] = vzero;

    for (int ks = 0; ks < Nn / 32; ++ks) {
      const int kb = ks * 32;
      if (kb + 32 < Nn)
        __builtin_prefetch(&Wp[(size_t)(kb + 32 + half * 16) * Dc + wave * 32], 0, 1);
      v16bf afr[4];  // contiguous bf16 runs -> ds_load_b128; shared by both nt
      #pragma unroll
      for (int r = 0; r < 4; ++r) {
        #pragma unroll
        for (int j = 0; j < 16; ++j) {
          const int kk = kb + ((j < 8) ? j : j + 8) + half * 8;
          afr[r][j] = sAct[(r * 16 + l16) * SAS + kk];
        }
      }
      #pragma unroll
      for (int nt = 0; nt < 2; ++nt) {
        const int nb = (wave * 2 + nt) * 16;
        v16bf bfr;
        #pragma unroll
        for (int j = 0; j < 16; ++j) {
          const size_t kk = (size_t)(kb + ((j < 8) ? j : j + 8) + half * 8);
          bfr[j] = (__bf16)Wp[kk * Dc + nb + l16];
        }
        #pragma unroll
        for (int r = 0; r < 4; ++r)
          acc[nt][r] = __builtin_amdgcn_wmma_f32_16x16x32_bf16(
              false, afr[r], false, bfr, (short)0, acc[nt][r], false, false);
      }
    }
    #pragma unroll
    for (int nt = 0; nt < 2; ++nt) {
      const int nb = (wave * 2 + nt) * 16;
      #pragma unroll
      for (int r = 0; r < 4; ++r) {
        #pragma unroll
        for (int v = 0; v < 8; ++v) {
          const int b = r * 16 + v + 8 * half;
          const int d = nb + l16;
          const float p = acc[nt][r][v];
          const size_t o = (size_t)b * Cc * Dc + (size_t)c * Dc + d;
          out_pred[o] = p;
          out_err[o]  = x_input[b * Dc + d] - p;
        }
      }
    }
  }
}

extern "C" void kernel_launch(void* const* d_in, const int* in_sizes, int n_in,
                              void* d_out, int out_size, void* d_ws, size_t ws_size,
                              hipStream_t stream) {
  (void)in_sizes; (void)n_in; (void)out_size; (void)d_ws; (void)ws_size;
  const float* x_input   = (const float*)d_in[0];
  const float* x_context = (const float*)d_in[1];
  const float* W_ff      = (const float*)d_in[2];
  const float* W_ctx     = (const float*)d_in[3];
  const float* W_pred    = (const float*)d_in[4];
  const float* bias      = (const float*)d_in[5];
  const float* avg       = (const float*)d_in[6];

  float* out_act  = (float*)d_out;                               // [B,C,N]
  float* out_pred = out_act  + (size_t)Bc * Cc * Nn;             // [B,C,D]
  float* out_err  = out_pred + (size_t)Bc * Cc * Dc;             // [B,C,D]

  // staged inputs (aliased by bf16 activations) + drive + boost  ~= 198 KB
  const size_t smem = (size_t)(2 * Bc * SXS) * sizeof(__bf16)
                    + (size_t)(Bc * DRS) * sizeof(float)
                    + (size_t)Nn * sizeof(float);

  cortical_region_kernel<<<dim3(Cc), dim3(256), smem, stream>>>(
      x_input, x_context, W_ff, W_ctx, W_pred, bias, avg,
      out_act, out_pred, out_err);
}
